// Prompt_60507499266896
// MI455X (gfx1250) — compile-verified, matrix operand order checked by reading, not run on hardware
//
#include <hip/hip_runtime.h>
#include <hip/hip_bf16.h>

#define B_    1024
#define S_    64
#define D_    768
#define POOL_ 30
#define TOPK_ 4
#define SEQ_  68
#define CT_   64
#define LDSW  776   // padded LDS row stride (bf16 elems): 776*2 bytes = 16B-aligned, bank-staggered

typedef __bf16 bf16;
typedef __attribute__((ext_vector_type(16))) __bf16 v16bf;
typedef __attribute__((ext_vector_type(8)))  float  v8f;
typedef __attribute__((ext_vector_type(2)))  float  f32x2;
typedef unsigned int u32;
typedef __attribute__((ext_vector_type(4))) u32 u32x4;

union Frag { v16bf v; u32x4 q[2]; };

__device__ inline float wave_sum(float v) {
#pragma unroll
    for (int m = 16; m >= 1; m >>= 1) v += __shfl_xor(v, m, 32);
    return v;
}

// ---------------------------------------------------------------- x_mean (B,D)
__global__ void __launch_bounds__(256) k_xmean(const float* __restrict__ x,
                                               float* __restrict__ xm) {
    int lin = blockIdx.x * 256 + threadIdx.x;
    if (lin >= B_ * D_) return;
    int b = lin / D_, d = lin % D_;
    const float* p = x + (size_t)b * S_ * D_ + d;
    float s = 0.f;
    for (int i = 0; i < S_; ++i) s += p[(size_t)i * D_];
    xm[lin] = s * (1.f / S_);
}

// ------------------------------------------------------- prompt_norm (POOL,D)
__global__ void __launch_bounds__(256) k_pnorm(const float* __restrict__ prompt,
                                               float* __restrict__ pn) {
    __shared__ float sw[8];
    const int r = blockIdx.x, t = threadIdx.x;
    float ps = 0.f;
    for (int d = t; d < D_; d += 256) { float v = prompt[r * D_ + d]; ps += v * v; }
    ps = wave_sum(ps);
    if ((t & 31) == 0) sw[t >> 5] = ps;
    __syncthreads();
    float ss = 0.f;
#pragma unroll
    for (int i = 0; i < 8; ++i) ss += sw[i];
    float rn = rsqrtf(fmaxf(ss, 1e-12f));
    for (int d = t; d < D_; d += 256) pn[r * D_ + d] = prompt[r * D_ + d] * rn;
}

// ----------------------------------- similarity, top-4, reduce_sim (per batch)
__global__ void __launch_bounds__(256) k_select(const float* __restrict__ xm,
                                                const float* __restrict__ pn,
                                                float* __restrict__ out_sim,
                                                float* __restrict__ out_idxf,
                                                float* __restrict__ out_rsim,
                                                int* __restrict__ widx) {
    __shared__ float sxn[D_];
    __shared__ float sw[8];
    __shared__ float ssim[POOL_];
    const int b = blockIdx.x, t = threadIdx.x, lane = t & 31, wid = t >> 5;

    float ps = 0.f;
    for (int d = t; d < D_; d += 256) { float v = xm[(size_t)b * D_ + d]; sxn[d] = v; ps += v * v; }
    ps = wave_sum(ps);
    if (lane == 0) sw[wid] = ps;
    __syncthreads();
    float ss = 0.f;
#pragma unroll
    for (int i = 0; i < 8; ++i) ss += sw[i];
    float rn = rsqrtf(fmaxf(ss, 1e-12f));
    for (int d = t; d < D_; d += 256) sxn[d] *= rn;
    __syncthreads();

    for (int j = wid; j < POOL_; j += 8) {
        float p = 0.f;
        for (int k = lane; k < D_; k += 32) p += sxn[k] * pn[j * D_ + k];
        p = wave_sum(p);
        if (lane == 0) ssim[j] = p;
    }
    __syncthreads();

    if (t == 0) {
        bool used[POOL_] = {};
        float rsum = 0.f;
        for (int k = 0; k < TOPK_; ++k) {
            float best = -3.4e38f; int bi = 0;
            for (int j = 0; j < POOL_; ++j)
                if (!used[j] && ssim[j] > best) { best = ssim[j]; bi = j; }
            used[bi] = true;
            widx[b * 4 + k] = bi;
            out_idxf[b * 4 + k] = (float)bi;
            rsum += best;
        }
        atomicAdd(out_rsim, rsum * (1.f / (float)B_));
    }
    if (t < POOL_) out_sim[(size_t)b * POOL_ + t] = ssim[t];
}

// --------------- pack w2a/w2b (768x768 f32) into WMMA-B bf16 fragment order
// block = (w, kt, nt); lane's 16 values contiguous:
//   lanes 0-15 (col n0+lane): K {k0..k0+7, k0+16..k0+23}
//   lanes 16-31 (col n0+l-16): K {k0+8..k0+15, k0+24..k0+31}
__global__ void __launch_bounds__(32) k_pack(const float* __restrict__ w2a,
                                             const float* __restrict__ w2b,
                                             bf16* __restrict__ pa,
                                             bf16* __restrict__ pb) {
    int blk = blockIdx.x;
    const int lane = threadIdx.x;
    const float* src; bf16* dst;
    if (blk < 1152) { src = w2a; dst = pa; } else { src = w2b; dst = pb; blk -= 1152; }
    const int kt = blk / 48, nt = blk % 48;
    const int r = lane & 15, half = lane >> 4;
    const int n = nt * 16 + r;
    const int kb = kt * 32 + half * 8;
    bf16* o = dst + (size_t)(kt * 48 + nt) * 512 + lane * 16;
#pragma unroll
    for (int i = 0; i < 16; ++i) {
        int k = kb + (i < 8 ? i : 8 + i);
        o[i] = (bf16)src[(size_t)k * D_ + n];
    }
}

// ------------------------------------------- GEMM 64x768 = A(64x768) * W(768x768)
// A in LDS (bf16, LDSW stride); W packed in global; bias+ReLU epilogue -> Dst LDS.
// Wave owns ALL 4 M-tiles x 3 N-tiles per pass (12 accumulators) with the K-loop
// outermost, so each B fragment (global) feeds 4 WMMAs and each A fragment (LDS)
// feeds 3 WMMAs: 0.5 global_b128 + 0.67 ds_b128 per WMMA.
__device__ inline void gemm_64x768(const bf16* __restrict__ Asrc,
                                   const bf16* __restrict__ Wpk,
                                   const float* __restrict__ bias,
                                   bf16* __restrict__ Dst,
                                   int lane, int wid) {
    const int r = lane & 15, half = lane >> 4;
#pragma unroll 1
    for (int pass = 0; pass < 2; ++pass) {
        int nts[3];
#pragma unroll
        for (int j = 0; j < 3; ++j) nts[j] = wid + 8 * (3 * pass + j);   // wave-uniform

        v8f acc[4][3];
#pragma unroll
        for (int mt = 0; mt < 4; ++mt)
#pragma unroll
            for (int j = 0; j < 3; ++j) acc[mt][j] = (v8f){};

#pragma unroll 1
        for (int kt = 0; kt < 24; ++kt) {
            Frag Af[4];
#pragma unroll
            for (int mt = 0; mt < 4; ++mt) {
                const bf16* ap = Asrc + (mt * 16 + r) * LDSW + kt * 32 + half * 8;
                Af[mt].q[0] = *(const u32x4*)(ap);        // K {base..base+7}
                Af[mt].q[1] = *(const u32x4*)(ap + 16);   // K {base+16..base+23}
            }
            Frag Bf[3];
#pragma unroll
            for (int j = 0; j < 3; ++j) {
                const bf16* bq = Wpk + (size_t)(kt * 48 + nts[j]) * 512 + lane * 16;
                Bf[j].q[0] = *(const u32x4*)(bq);
                Bf[j].q[1] = *(const u32x4*)(bq + 8);
            }
#pragma unroll
            for (int mt = 0; mt < 4; ++mt)
#pragma unroll
                for (int j = 0; j < 3; ++j)
                    acc[mt][j] = __builtin_amdgcn_wmma_f32_16x16x32_bf16(
                        false, Af[mt].v, false, Bf[j].v, (short)0, acc[mt][j], false, false);
        }

#pragma unroll
        for (int j = 0; j < 3; ++j) {
            const int n = nts[j] * 16 + r;
            const float bs = bias[n];
#pragma unroll
            for (int mt = 0; mt < 4; ++mt) {
#pragma unroll
                for (int i = 0; i < 8; ++i) {             // C: vgpr i -> M=i / M=8+i
                    float vv = fmaxf(acc[mt][j][i] + bs, 0.f);
                    Dst[(mt * 16 + half * 8 + i) * LDSW + n] = (bf16)vv;
                }
            }
        }
    }
}

// ---------------- row LayerNorm over 768, wave32 butterfly, write back bf16
__device__ inline void ln_rows_bf(bf16* Buf, const float* __restrict__ g,
                                  const float* __restrict__ be, int lane, int wid) {
    for (int row = wid; row < CT_; row += 8) {
        float vals[24];
        float s = 0.f, sq = 0.f;
#pragma unroll
        for (int k = 0; k < 24; ++k) {
            float v = (float)Buf[row * LDSW + lane + 32 * k];
            vals[k] = v; s += v; sq += v * v;
        }
        s = wave_sum(s); sq = wave_sum(sq);
        float m = s * (1.f / 768.f);
        float var = fmaxf(sq * (1.f / 768.f) - m * m, 0.f);
        float rs = rsqrtf(var + 1e-5f);
#pragma unroll
        for (int k = 0; k < 24; ++k) {
            int n = lane + 32 * k;
            Buf[row * LDSW + n] = (bf16)((vals[k] - m) * rs * g[n] + be[n]);
        }
    }
}

__device__ inline void ln_rows_out(const bf16* Buf, const float* __restrict__ g,
                                   const float* __restrict__ be,
                                   float* __restrict__ out, int lane, int wid) {
    for (int row = wid; row < CT_; row += 8) {
        float vals[24];
        float s = 0.f, sq = 0.f;
#pragma unroll
        for (int k = 0; k < 24; ++k) {
            float v = (float)Buf[row * LDSW + lane + 32 * k];
            vals[k] = v; s += v; sq += v * v;
        }
        s = wave_sum(s); sq = wave_sum(sq);
        float m = s * (1.f / 768.f);
        float var = fmaxf(sq * (1.f / 768.f) - m * m, 0.f);
        float rs = rsqrtf(var + 1e-5f);
#pragma unroll
        for (int k = 0; k < 24; ++k) {
            int n = lane + 32 * k;
            out[(size_t)row * D_ + n] = (vals[k] - m) * rs * g[n] + be[n];
        }
    }
}

// ------------------------------------------------ fused mixer (one block per b)
__global__ void __launch_bounds__(256) k_main(
    const float* __restrict__ x_embed, const float* __restrict__ prompt,
    const float* __restrict__ w1a, const float* __restrict__ b1a,
    const float* __restrict__ g1a, const float* __restrict__ be1a,
    const float* __restrict__ w1b, const float* __restrict__ b1b,
    const float* __restrict__ g1b, const float* __restrict__ be1b,
    const float* __restrict__ b2a, const float* __restrict__ g2a, const float* __restrict__ be2a,
    const float* __restrict__ b2b, const float* __restrict__ g2b, const float* __restrict__ be2b,
    const bf16* __restrict__ pw2a, const bf16* __restrict__ pw2b,
    const int* __restrict__ idx, float* __restrict__ out) {
    extern __shared__ char smem[];
    bf16* Hbuf = (bf16*)smem;                     // 64 x LDSW bf16
    bf16* Tbuf = (bf16*)(smem + 64 * LDSW * 2);   // 64 x LDSW bf16
    float* Wst   = (float*)Tbuf;                  // phase-1 overlay (weights staging)
    float* Lw1a  = Wst;            // 68*64
    float* Lb1a  = Wst + 4352;
    float* Lw1b  = Wst + 4416;     // 64*64
    float* Lb1b  = Wst + 8512;
    float* Lg1a  = Wst + 8576;
    float* Lbe1a = Wst + 8640;
    float* Lg1b  = Wst + 8704;
    float* Lbe1b = Wst + 8768;
    __shared__ int s_idx[4];

    const int b = blockIdx.x;
    const int t = threadIdx.x;
    const int lane = t & 31, wid = t >> 5;

    // phase 0: stage mlp_1 weights + idx
    for (int i = t; i < SEQ_ * CT_; i += 256) Lw1a[i] = w1a[i];
    for (int i = t; i < CT_ * CT_; i += 256)  Lw1b[i] = w1b[i];
    if (t < CT_) {
        Lb1a[t] = b1a[t]; Lb1b[t] = b1b[t];
        Lg1a[t] = g1a[t]; Lbe1a[t] = be1a[t];
        Lg1b[t] = g1b[t]; Lbe1b[t] = be1b[t];
    }
    if (t < 4) s_idx[t] = idx[b * 4 + t];
    __syncthreads();

    // phase 1: mlp_1 — each thread owns channel d: row of x^T (len 68) -> CT -> CT.
    // Packed f32 math (V_PK_FMA_F32): 32 float2 accumulators per layer.
    const f32x2 zero2 = {0.f, 0.f};
    for (int p = 0; p < 3; ++p) {
        const int d = p * 256 + t;
        f32x2 a[32];
        const f32x2* pb1a = (const f32x2*)Lb1a;
#pragma unroll
        for (int j = 0; j < 32; ++j) a[j] = pb1a[j];
        const float* xb = x_embed + (size_t)b * S_ * D_ + d;
        for (int s = 0; s < SEQ_; ++s) {
            float xv = (s < 4) ? prompt[(size_t)s_idx[s] * D_ + d]
                               : xb[(size_t)(s - 4) * D_];
            f32x2 xv2 = {xv, xv};
            const f32x2* wr = (const f32x2*)(Lw1a + s * CT_);
#pragma unroll
            for (int j = 0; j < 32; ++j) a[j] = __builtin_elementwise_fma(xv2, wr[j], a[j]);
        }
        // relu + LN(g1a,be1a)
        f32x2 ms = zero2;
#pragma unroll
        for (int j = 0; j < 32; ++j) { a[j] = __builtin_elementwise_max(a[j], zero2); ms += a[j]; }
        float m = (ms[0] + ms[1]) * (1.f / CT_);
        f32x2 m2 = {m, m};
        f32x2 vs = zero2;
#pragma unroll
        for (int j = 0; j < 32; ++j) { f32x2 u = a[j] - m2; vs = __builtin_elementwise_fma(u, u, vs); }
        float rs = rsqrtf((vs[0] + vs[1]) * (1.f / CT_) + 1e-5f);
        f32x2 rs2 = {rs, rs};
        const f32x2* pg1a = (const f32x2*)Lg1a;
        const f32x2* pbe1a = (const f32x2*)Lbe1a;
#pragma unroll
        for (int j = 0; j < 32; ++j) {
            f32x2 u = (a[j] - m2) * rs2;
            a[j] = __builtin_elementwise_fma(u, pg1a[j], pbe1a[j]);
        }

        // layer 2
        f32x2 c[32];
        const f32x2* pb1b = (const f32x2*)Lb1b;
#pragma unroll
        for (int j = 0; j < 32; ++j) c[j] = pb1b[j];
#pragma unroll
        for (int s2 = 0; s2 < CT_; ++s2) {
            const float av = a[s2 >> 1][s2 & 1];
            f32x2 av2 = {av, av};
            const f32x2* wr = (const f32x2*)(Lw1b + s2 * CT_);
#pragma unroll
            for (int j = 0; j < 32; ++j) c[j] = __builtin_elementwise_fma(av2, wr[j], c[j]);
        }
        ms = zero2;
#pragma unroll
        for (int j = 0; j < 32; ++j) { c[j] = __builtin_elementwise_max(c[j], zero2); ms += c[j]; }
        m = (ms[0] + ms[1]) * (1.f / CT_);
        m2 = (f32x2){m, m};
        vs = zero2;
#pragma unroll
        for (int j = 0; j < 32; ++j) { f32x2 u = c[j] - m2; vs = __builtin_elementwise_fma(u, u, vs); }
        rs = rsqrtf((vs[0] + vs[1]) * (1.f / CT_) + 1e-5f);
        rs2 = (f32x2){rs, rs};
        const f32x2* pg1b = (const f32x2*)Lg1b;
        const f32x2* pbe1b = (const f32x2*)Lbe1b;
#pragma unroll
        for (int j = 0; j < 32; ++j) {                 // transpose-store: h[2j][d], h[2j+1][d]
            f32x2 u = (c[j] - m2) * rs2;
            f32x2 o = __builtin_elementwise_fma(u, pg1b[j], pbe1b[j]);
            Hbuf[(2 * j) * LDSW + d]     = (bf16)o[0];
            Hbuf[(2 * j + 1) * LDSW + d] = (bf16)o[1];
        }
    }
    __syncthreads();

    // phase 2/3: z1 = LN(relu(h @ w2a + b2a))
    gemm_64x768(Hbuf, pw2a, b2a, Tbuf, lane, wid);
    __syncthreads();
    ln_rows_bf(Tbuf, g2a, be2a, lane, wid);
    __syncthreads();

    // phase 4/5: out = LN(relu(z1 @ w2b + b2b))
    gemm_64x768(Tbuf, pw2b, b2b, Hbuf, lane, wid);
    __syncthreads();
    ln_rows_out(Hbuf, g2b, be2b, out + (size_t)b * CT_ * D_, lane, wid);
}

extern "C" void kernel_launch(void* const* d_in, const int* in_sizes, int n_in,
                              void* d_out, int out_size, void* d_ws, size_t ws_size,
                              hipStream_t stream) {
    (void)in_sizes; (void)n_in; (void)out_size; (void)ws_size;
    const float* x_embed = (const float*)d_in[0];
    const float* prompt  = (const float*)d_in[1];
    const float* w1a  = (const float*)d_in[2];  const float* b1a  = (const float*)d_in[3];
    const float* g1a  = (const float*)d_in[4];  const float* be1a = (const float*)d_in[5];
    const float* w1b  = (const float*)d_in[6];  const float* b1b  = (const float*)d_in[7];
    const float* g1b  = (const float*)d_in[8];  const float* be1b = (const float*)d_in[9];
    const float* w2a  = (const float*)d_in[10]; const float* b2a  = (const float*)d_in[11];
    const float* g2a  = (const float*)d_in[12]; const float* be2a = (const float*)d_in[13];
    const float* w2b  = (const float*)d_in[14]; const float* b2b  = (const float*)d_in[15];
    const float* g2b  = (const float*)d_in[16]; const float* be2b = (const float*)d_in[17];

    float* out = (float*)d_out;
    const size_t O_RSIM = (size_t)B_ * CT_ * D_;     // 50,331,648
    const size_t O_SIM  = O_RSIM + 1;
    const size_t O_IDX  = O_SIM + (size_t)B_ * POOL_;

    char* ws = (char*)d_ws;
    float* ws_xmean = (float*)ws;                                // B*D f32
    float* ws_pnorm = (float*)(ws + 12582912);                   // 30*768 f32
    int*   ws_idx   = (int*)  (ws + 12582912 + 92160);           // B*4 i32
    bf16*  ws_w2a   = (bf16*) (ws + 12582912 + 92160 + 16384);   // 1152*512 bf16
    bf16*  ws_w2b   = ws_w2a + (size_t)1152 * 512;

    hipError_t err;
    err = hipMemsetAsync(out + O_RSIM, 0, sizeof(float), stream);
    (void)err;
    k_pnorm<<<POOL_, 256, 0, stream>>>(prompt, ws_pnorm);
    k_xmean<<<(B_ * D_ + 255) / 256, 256, 0, stream>>>(x_embed, ws_xmean);
    k_pack<<<2304, 32, 0, stream>>>(w2a, w2b, ws_w2a, ws_w2b);
    k_select<<<B_, 256, 0, stream>>>(ws_xmean, ws_pnorm, out + O_SIM, out + O_IDX,
                                     out + O_RSIM, ws_idx);

    const size_t DynLDS = (size_t)2 * 64 * LDSW * 2;  // 198,656 B (< 320KB WGP LDS)
    err = hipFuncSetAttribute(reinterpret_cast<const void*>(k_main),
                              hipFuncAttributeMaxDynamicSharedMemorySize, (int)DynLDS);
    (void)err;
    k_main<<<B_, 256, DynLDS, stream>>>(x_embed, prompt,
                                        w1a, b1a, g1a, be1a, w1b, b1b, g1b, be1b,
                                        b2a, g2a, be2a, b2b, g2b, be2b,
                                        ws_w2a, ws_w2b, ws_idx, out);
}
